// PointNetPlusPlus_45483703665264
// MI455X (gfx1250) — matrix operand order repaired and checked by reading, not compiled, not run on hardware
//
#include <hip/hip_runtime.h>
#include <stddef.h>
#include <stdint.h>

// ---------------------------------------------------------------------------
// PointNet++ forward for MI455X (gfx1250, wave32).
// All dense linear layers run through v_wmma_f32_16x16x32_bf16 (bf16 A/B,
// f32 accumulate) with weights pre-transposed to bf16 [N][Kpad] so the GEMM
// inner loop is pure b128 global->LDS staging + WMMA.  Irregular ops (FPS,
// radius/top-K query, knn-interp, masked BN, masked max) are block-per-query
// LDS-reduction kernels.
// ---------------------------------------------------------------------------

typedef __attribute__((ext_vector_type(16))) __bf16 v16bf;
typedef __attribute__((ext_vector_type(8)))  float  v8f;

#define FLT_MAX_C 3.402823466e+38f
#define INT_MAX_C 0x7fffffff

// ======================== weight pre-transpose ============================
// WT[n][k] = (bf16)W[k*N + n]  for k < Kdim, else 0   (Kpad = roundup32(Kdim))
__global__ __launch_bounds__(256)
void transpose_w_kernel(const float* __restrict__ W, int Kdim, int N, int Kpad,
                        __bf16* __restrict__ WT)
{
    const size_t total  = (size_t)N * Kpad;
    const size_t stride = (size_t)gridDim.x * blockDim.x;
    for (size_t e = (size_t)blockIdx.x * blockDim.x + threadIdx.x;
         e < total; e += stride) {
        const int n = (int)(e / Kpad);
        const int k = (int)(e % Kpad);
        const float v = (k < Kdim) ? W[(size_t)k * N + n] : 0.f;
        WT[e] = (__bf16)v;
    }
}

// ======================== WMMA GEMM =======================================
// C[M x N] = A[M x Kpad](bf16, zero-padded) * WT[N x Kpad](bf16, transposed)
//            + bias
// Block: 256 threads = 8 waves; block tile 64(M) x 64(N).
// Waves in 4(M) x 2(N); each wave computes 16 x 32 via two WMMA accumulators.
__global__ __launch_bounds__(256)
void wmma_gemm_kernel(const __bf16* __restrict__ A, int Kpad,
                      const __bf16* __restrict__ WT,
                      const float* __restrict__ bias,
                      float* __restrict__ Cf, __bf16* __restrict__ Cb,
                      int ldc, int M)
{
    __shared__ __bf16 As[64][32];     // 4 KB
    __shared__ __bf16 WsT[64][32];    // 4 KB, [n][k]

    const int t    = threadIdx.x;
    const int bm   = blockIdx.x * 64;
    const int bn   = blockIdx.y * 64;
    const int lane = t & 31;
    const int wid  = t >> 5;
    const int wm   = wid & 3;         // 4 wave rows (16 M each)
    const int wn   = wid >> 2;        // 2 wave cols (32 N each)
    const int half = lane >> 4;
    const int l16  = lane & 15;

    // staging maps: 256 threads x 8 bf16 = 64 x 32 tile
    const int ar = t >> 2;
    const int ac = (t & 3) * 8;
    const int wr = t >> 2;            // n within tile
    const int wc = (t & 3) * 8;       // k within tile

    v8f acc0 = {};
    v8f acc1 = {};

    const __bf16* aptr = A  + (size_t)(bm + ar) * Kpad + ac;
    const __bf16* wptr = WT + (size_t)(bn + wr) * Kpad + wc;

    for (int k0 = 0; k0 < Kpad; k0 += 32) {
        // --- stage A + W tiles: one 16B global load -> one 16B LDS store each
        *(uint4*)(&As[ar][ac])  = *(const uint4*)(aptr + k0);
        *(uint4*)(&WsT[wr][wc]) = *(const uint4*)(wptr + k0);
        __builtin_prefetch(aptr + k0 + 32, 0, 1);   // speculative: dropped if OOB
        __syncthreads();

        // --- build fragments (A 16x32 / B 32x16, contiguous k = 16*half + s in both)
        v16bf afrag, bfrag0, bfrag1;
        {
            union { uint4 u[2]; v16bf v; } ua;
            const __bf16* ap = &As[16 * wm + l16][half * 16];
            ua.u[0] = *(const uint4*)(ap);        // K = base .. base+7
            ua.u[1] = *(const uint4*)(ap + 8);    // K = base+8 .. base+15
            afrag = ua.v;
        }
        {
            union { uint4 u[2]; v16bf v; } ub;
            const __bf16* bp = &WsT[32 * wn + l16][half * 16];
            ub.u[0] = *(const uint4*)(bp);        // K = base .. base+7
            ub.u[1] = *(const uint4*)(bp + 8);    // K = base+8 .. base+15
            bfrag0 = ub.v;
        }
        {
            union { uint4 u[2]; v16bf v; } ub;
            const __bf16* bp = &WsT[32 * wn + 16 + l16][half * 16];
            ub.u[0] = *(const uint4*)(bp);
            ub.u[1] = *(const uint4*)(bp + 8);
            bfrag1 = ub.v;
        }

        acc0 = __builtin_amdgcn_wmma_f32_16x16x32_bf16(
                   false, afrag, false, bfrag0, (short)0, acc0, false, false);
        acc1 = __builtin_amdgcn_wmma_f32_16x16x32_bf16(
                   false, afrag, false, bfrag1, (short)0, acc1, false, false);
        __syncthreads();
    }

    // --- epilogue: bias add + store (f32 or bf16)
    const int col0 = bn + 32 * wn + l16;
    const int col1 = col0 + 16;
    const float bv0 = bias ? bias[col0] : 0.f;
    const float bv1 = bias ? bias[col1] : 0.f;
#pragma unroll
    for (int r = 0; r < 8; ++r) {
        const int row = bm + 16 * wm + r + half * 8;
        if (row < M) {
            const float v0 = acc0[r] + bv0;
            const float v1 = acc1[r] + bv1;
            if (Cf) {
                Cf[(size_t)row * ldc + col0] = v0;
                Cf[(size_t)row * ldc + col1] = v1;
            } else {
                Cb[(size_t)row * ldc + col0] = (__bf16)v0;
                Cb[(size_t)row * ldc + col1] = (__bf16)v1;
            }
        }
    }
}

// ======================== small utility kernels ===========================
__global__ void zero_f32_kernel(float* p, int n)
{
    int i = blockIdx.x * blockDim.x + threadIdx.x;
    if (i < n) p[i] = 0.f;
}

// Masked BatchNorm statistics: stats[0..N-1]=sum, [N..2N-1]=sumsq, [2N]=count
__global__ __launch_bounds__(256)
void bn_stats_kernel(const __bf16* __restrict__ H, int ldh,
                     const int* __restrict__ mask, int M, int N,
                     float* __restrict__ stats)
{
    const int t = threadIdx.x;
    float s = 0.f, sq = 0.f, cnt = 0.f;
    for (int r = blockIdx.x; r < M; r += gridDim.x) {
        const float w = mask ? (float)mask[r] : 1.f;
        if (t < N) {
            const float h = (float)H[(size_t)r * ldh + t];
            s  += h * w;
            sq += h * h * w;
        }
        if (t == 0) cnt += w;
    }
    if (t < N) {
        atomicAdd(&stats[t], s);
        atomicAdd(&stats[N + t], sq);
    }
    if (t == 0) atomicAdd(&stats[2 * N], cnt);
}

// In-place normalize + ReLU on bf16 activations
__global__ __launch_bounds__(256)
void bn_norm_relu_kernel(__bf16* H, int ldh, int M, int N,
                         const float* __restrict__ stats)
{
    const size_t total  = (size_t)M * N;
    const size_t stride = (size_t)gridDim.x * blockDim.x;
    const float cnt = fmaxf(stats[2 * N], 1.f);
    for (size_t e = (size_t)blockIdx.x * blockDim.x + threadIdx.x;
         e < total; e += stride) {
        const int    f = (int)(e % N);
        const size_t r = e / N;
        const float mean = stats[f] / cnt;
        const float var  = stats[N + f] / cnt - mean * mean;
        float h = (float)H[r * ldh + f];
        h = (h - mean) * rsqrtf(var + 1e-5f);
        H[r * ldh + f] = (__bf16)fmaxf(h, 0.f);
    }
}

// Masked max over K neighbors -> f32 [Q x N]; one block per query
__global__ __launch_bounds__(256)
void masked_max_kernel(const __bf16* __restrict__ H, int ldh,
                       const int* __restrict__ mask, int Kn, int N,
                       float* __restrict__ out)
{
    const int q = blockIdx.x;
    const int f = threadIdx.x;
    if (f >= N) return;
    float best = -FLT_MAX_C;
    for (int k = 0; k < Kn; ++k) {
        if (mask[(size_t)q * Kn + k]) {
            const float h = (float)H[((size_t)q * Kn + k) * ldh + f];
            best = fmaxf(best, h);
        }
    }
    out[(size_t)q * N + f] = best;
}

// ======================== input-building kernels ==========================
// x0 input: [emb[tok] | prompt] padded to ldm, bf16
__global__ __launch_bounds__(256)
void build_x0_in_kernel(const int* __restrict__ tok, const int* __restrict__ index,
                        const float* __restrict__ emb, const float* __restrict__ pemb,
                        int n, int ldm, __bf16* __restrict__ A, int Bn)
{
    const size_t total  = (size_t)Bn * n * ldm;
    const size_t stride = (size_t)gridDim.x * blockDim.x;
    for (size_t e = (size_t)blockIdx.x * blockDim.x + threadIdx.x;
         e < total; e += stride) {
        const int    c   = (int)(e % ldm);
        const size_t row = e / ldm;
        const int    b   = (int)(row / n);
        float v = 0.f;
        if (c < 128) {
            v = emb[(size_t)tok[row] * 128 + c];
        } else if (c < 136) {
            int pi = index[b] / 50;
            pi = pi < 0 ? 0 : (pi > 20 ? 20 : pi);
            v = pemb[pi * 8 + (c - 128)];
        }
        A[e] = (__bf16)v;
    }
}

// [H | prompt] padded to ldm, bf16
__global__ __launch_bounds__(256)
void concat_prompt_kernel(const float* __restrict__ H, int F,
                          const int* __restrict__ index, const float* __restrict__ pemb,
                          int rowsPerB, int ldm, __bf16* __restrict__ A, int Bn)
{
    const size_t total  = (size_t)Bn * rowsPerB * ldm;
    const size_t stride = (size_t)gridDim.x * blockDim.x;
    for (size_t e = (size_t)blockIdx.x * blockDim.x + threadIdx.x;
         e < total; e += stride) {
        const int    c   = (int)(e % ldm);
        const size_t row = e / ldm;
        const int    b   = (int)(row / rowsPerB);
        float v = 0.f;
        if (c < F) {
            v = H[row * F + c];
        } else if (c < F + 8) {
            int pi = index[b] / 50;
            pi = pi < 0 ? 0 : (pi > 20 ? 20 : pi);
            v = pemb[pi * 8 + (c - F)];
        }
        A[e] = (__bf16)v;
    }
}

// [X | Y] -> bf16 (ldm = Fx + Fy, multiple of 32)
__global__ __launch_bounds__(256)
void concat2_kernel(const float* __restrict__ X, int Fx,
                    const float* __restrict__ Y, int Fy,
                    int rows, int ldm, __bf16* __restrict__ A)
{
    const size_t total  = (size_t)rows * ldm;
    const size_t stride = (size_t)gridDim.x * blockDim.x;
    for (size_t e = (size_t)blockIdx.x * blockDim.x + threadIdx.x;
         e < total; e += stride) {
        const int    c   = (int)(e % ldm);
        const size_t row = e / ldm;
        const float v = (c < Fx) ? X[row * Fx + c] : Y[row * Fy + (c - Fx)];
        A[e] = (__bf16)v;
    }
}

// edge message matrix: [x[nbr] | pos[nbr]-pos_q | 0-pad], bf16
__global__ __launch_bounds__(256)
void gather_msg_kernel(const float* __restrict__ feat, int F,
                       const float* __restrict__ pos, const float* __restrict__ posq,
                       const int* __restrict__ nbr,
                       int n, int m, int Kn, int ldm, __bf16* __restrict__ A, int Bn)
{
    const size_t total  = (size_t)Bn * m * Kn * ldm;
    const size_t stride = (size_t)gridDim.x * blockDim.x;
    for (size_t e = (size_t)blockIdx.x * blockDim.x + threadIdx.x;
         e < total; e += stride) {
        const int    c   = (int)(e % ldm);
        const size_t row = e / ldm;
        const size_t qq  = row / Kn;
        const int    q   = (int)(qq % m);
        const int    b   = (int)(qq / m);
        const int    j   = nbr[row];
        float v = 0.f;
        if (c < F) {
            v = feat[((size_t)b * n + j) * F + c];
        } else if (c < F + 3) {
            const int d = c - F;
            v = pos[((size_t)b * n + j) * 3 + d] - posq[((size_t)b * m + q) * 3 + d];
        }
        A[e] = (__bf16)v;
    }
}

// ======================== FPS (block per batch) ===========================
__global__ __launch_bounds__(256)
void fps_kernel(const float* __restrict__ pos, int n, int m, int* __restrict__ idx)
{
    __shared__ float mind[2048];
    __shared__ float rv[256];
    __shared__ int   ri[256];
    __shared__ int   s_last;

    const int t = threadIdx.x;
    const int b = blockIdx.x;

    for (int i = t; i < n; i += 256) mind[i] = 1e10f;
    if (t == 0) { idx[b * m] = 0; s_last = 0; }
    __syncthreads();

    int last = 0;
    for (int it = 1; it < m; ++it) {
        const float lx = pos[((size_t)b * n + last) * 3 + 0];
        const float ly = pos[((size_t)b * n + last) * 3 + 1];
        const float lz = pos[((size_t)b * n + last) * 3 + 2];
        float bv = -FLT_MAX_C;
        int   bi = INT_MAX_C;
        for (int i = t; i < n; i += 256) {
            const float dx = pos[((size_t)b * n + i) * 3 + 0] - lx;
            const float dy = pos[((size_t)b * n + i) * 3 + 1] - ly;
            const float dz = pos[((size_t)b * n + i) * 3 + 2] - lz;
            const float d  = dx * dx + dy * dy + dz * dz;
            const float mn = fminf(mind[i], d);
            mind[i] = mn;
            if (mn > bv) { bv = mn; bi = i; }   // strict > keeps first index
        }
        rv[t] = bv; ri[t] = bi;
        __syncthreads();
        for (int s = 128; s > 0; s >>= 1) {
            if (t < s) {
                if (rv[t + s] > rv[t] || (rv[t + s] == rv[t] && ri[t + s] < ri[t])) {
                    rv[t] = rv[t + s]; ri[t] = ri[t + s];
                }
            }
            __syncthreads();
        }
        if (t == 0) { s_last = ri[0]; idx[b * m + it] = ri[0]; }
        __syncthreads();
        last = s_last;
    }
}

__global__ void gather_posq_kernel(const float* __restrict__ pos,
                                   const int* __restrict__ idx,
                                   int n, int m, int Bn, float* __restrict__ posq)
{
    const int tindex = blockIdx.x * blockDim.x + threadIdx.x;
    if (tindex >= Bn * m) return;
    const int b = tindex / m;
    const int j = idx[tindex];
#pragma unroll
    for (int d = 0; d < 3; ++d)
        posq[(size_t)tindex * 3 + d] = pos[((size_t)b * n + j) * 3 + d];
}

// ======================== nearest-K radius query ==========================
__global__ __launch_bounds__(256)
void knn_query_kernel(const float* __restrict__ pos, const float* __restrict__ posq,
                      int n, int m, int Kn, float r2,
                      int* __restrict__ nbr, int* __restrict__ mask)
{
    __shared__ float d2s[2048];
    __shared__ float rv[256];
    __shared__ int   ri[256];

    const int t  = threadIdx.x;
    const int bq = blockIdx.x;          // b*m + q
    const int b  = bq / m;

    const float qx = posq[(size_t)bq * 3 + 0];
    const float qy = posq[(size_t)bq * 3 + 1];
    const float qz = posq[(size_t)bq * 3 + 2];
    for (int i = t; i < n; i += 256) {
        const float dx = pos[((size_t)b * n + i) * 3 + 0] - qx;
        const float dy = pos[((size_t)b * n + i) * 3 + 1] - qy;
        const float dz = pos[((size_t)b * n + i) * 3 + 2] - qz;
        d2s[i] = dx * dx + dy * dy + dz * dz;
    }
    __syncthreads();

    for (int s = 0; s < Kn; ++s) {
        float bv = FLT_MAX_C;
        int   bi = INT_MAX_C;
        for (int i = t; i < n; i += 256) {
            const float d = d2s[i];
            if (d < bv) { bv = d; bi = i; }
        }
        rv[t] = bv; ri[t] = bi;
        __syncthreads();
        for (int st = 128; st > 0; st >>= 1) {
            if (t < st) {
                if (rv[t + st] < rv[t] || (rv[t + st] == rv[t] && ri[t + st] < ri[t])) {
                    rv[t] = rv[t + st]; ri[t] = ri[t + st];
                }
            }
            __syncthreads();
        }
        if (t == 0) {
            const int j = ri[0];
            nbr[(size_t)bq * Kn + s]  = j;
            mask[(size_t)bq * Kn + s] = (rv[0] <= r2) ? 1 : 0;
            d2s[j] = FLT_MAX_C;
        }
        __syncthreads();
    }
}

// ======================== knn interpolate (k=3) ===========================
__global__ __launch_bounds__(256)
void knn_interp3_kernel(const float* __restrict__ xc, const float* __restrict__ posc,
                        const float* __restrict__ posf, int nc, int nf, int F,
                        float* __restrict__ out)
{
    __shared__ float cd[768];
    __shared__ int   ci[768];
    __shared__ int   sidx[3];
    __shared__ float sw[3];
    __shared__ float swsum;

    const int t  = threadIdx.x;
    const int bi = blockIdx.x;           // b*nf + i
    const int b  = bi / nf;

    const float fx = posf[(size_t)bi * 3 + 0];
    const float fy = posf[(size_t)bi * 3 + 1];
    const float fz = posf[(size_t)bi * 3 + 2];

    float bd0 = FLT_MAX_C, bd1 = FLT_MAX_C, bd2 = FLT_MAX_C;
    int   bj0 = INT_MAX_C, bj1 = INT_MAX_C, bj2 = INT_MAX_C;
    for (int j = t; j < nc; j += 256) {
        const float dx = posc[((size_t)b * nc + j) * 3 + 0] - fx;
        const float dy = posc[((size_t)b * nc + j) * 3 + 1] - fy;
        const float dz = posc[((size_t)b * nc + j) * 3 + 2] - fz;
        const float d  = dx * dx + dy * dy + dz * dz;
        if (d < bd0)      { bd2 = bd1; bj2 = bj1; bd1 = bd0; bj1 = bj0; bd0 = d; bj0 = j; }
        else if (d < bd1) { bd2 = bd1; bj2 = bj1; bd1 = d; bj1 = j; }
        else if (d < bd2) { bd2 = d; bj2 = j; }
    }
    cd[t * 3 + 0] = bd0; ci[t * 3 + 0] = bj0;
    cd[t * 3 + 1] = bd1; ci[t * 3 + 1] = bj1;
    cd[t * 3 + 2] = bd2; ci[t * 3 + 2] = bj2;
    __syncthreads();

    if (t == 0) {
        float fd[3] = { FLT_MAX_C, FLT_MAX_C, FLT_MAX_C };
        int   fj[3] = { INT_MAX_C, INT_MAX_C, INT_MAX_C };
        for (int s = 0; s < 768; ++s) {
            const float d = cd[s];
            const int   j = ci[s];
            if (j == INT_MAX_C) continue;
            if (d < fd[0] || (d == fd[0] && j < fj[0])) {
                fd[2] = fd[1]; fj[2] = fj[1]; fd[1] = fd[0]; fj[1] = fj[0];
                fd[0] = d; fj[0] = j;
            } else if (d < fd[1] || (d == fd[1] && j < fj[1])) {
                fd[2] = fd[1]; fj[2] = fj[1]; fd[1] = d; fj[1] = j;
            } else if (d < fd[2] || (d == fd[2] && j < fj[2])) {
                fd[2] = d; fj[2] = j;
            }
        }
        float ws = 0.f;
#pragma unroll
        for (int s = 0; s < 3; ++s) {
            sw[s]   = 1.f / fmaxf(fd[s], 1e-16f);
            sidx[s] = fj[s];
            ws += sw[s];
        }
        swsum = ws;
    }
    __syncthreads();

    for (int f = t; f < F; f += 256) {
        float acc = 0.f;
#pragma unroll
        for (int s = 0; s < 3; ++s)
            acc += sw[s] * xc[((size_t)b * nc + sidx[s]) * F + f];
        out[(size_t)bi * F + f] = acc / swsum;
    }
}

// ======================== host launch ======================================
extern "C" void kernel_launch(void* const* d_in, const int* in_sizes, int n_in,
                              void* d_out, int out_size, void* d_ws, size_t ws_size,
                              hipStream_t stream)
{
    (void)in_sizes; (void)out_size; (void)ws_size;

    // ---- inputs (setup_inputs dict order, lists flattened to leaves) ----
    const int*   x_tok = (const int*)  d_in[0];
    const float* pos   = (const float*)d_in[1];
    const int*   index = (const int*)  d_in[3];
    const float* emb   = (const float*)d_in[4];
    const float* pemb  = (const float*)d_in[5];
    const float* W1 = (const float*)d_in[6],  *b1 = (const float*)d_in[7];
    const float* W2 = (const float*)d_in[8],  *b2 = (const float*)d_in[9];
    const float* W3 = (const float*)d_in[10], *b3 = (const float*)d_in[11];
    const float* W4 = (const float*)d_in[12], *b4 = (const float*)d_in[13];

    const float *sa1W[3], *sa1b[3], *sa2W[3], *sa2b[3];
    const float *fp2W[2], *fp2b[2], *fp1W[3], *fp1b[3];
    if (n_in >= 36) {
        for (int i = 0; i < 3; ++i) { sa1W[i] = (const float*)d_in[14 + i]; }
        for (int i = 0; i < 3; ++i) { sa1b[i] = (const float*)d_in[17 + i]; }
        for (int i = 0; i < 3; ++i) { sa2W[i] = (const float*)d_in[20 + i]; }
        for (int i = 0; i < 3; ++i) { sa2b[i] = (const float*)d_in[23 + i]; }
        for (int i = 0; i < 2; ++i) { fp2W[i] = (const float*)d_in[26 + i]; }
        for (int i = 0; i < 2; ++i) { fp2b[i] = (const float*)d_in[28 + i]; }
        for (int i = 0; i < 3; ++i) { fp1W[i] = (const float*)d_in[30 + i]; }
        for (int i = 0; i < 3; ++i) { fp1b[i] = (const float*)d_in[33 + i]; }
    } else {
        // fallback: each list passed as one concatenated buffer (n_in == 22)
        const float* p;
        p = (const float*)d_in[14]; sa1W[0] = p; sa1W[1] = p + 131 * 128; sa1W[2] = sa1W[1] + 128 * 128;
        p = (const float*)d_in[15]; sa1b[0] = p; sa1b[1] = p + 128;       sa1b[2] = p + 256;
        p = (const float*)d_in[16]; sa2W[0] = p; sa2W[1] = p + 259 * 256; sa2W[2] = sa2W[1] + 256 * 256;
        p = (const float*)d_in[17]; sa2b[0] = p; sa2b[1] = p + 256;       sa2b[2] = p + 512;
        p = (const float*)d_in[18]; fp2W[0] = p; fp2W[1] = p + 512 * 256;
        p = (const float*)d_in[19]; fp2b[0] = p; fp2b[1] = p + 256;
        p = (const float*)d_in[20]; fp1W[0] = p; fp1W[1] = p + 384 * 256; fp1W[2] = fp1W[1] + 256 * 256;
        p = (const float*)d_in[21]; fp1b[0] = p; fp1b[1] = p + 256;       fp1b[2] = fp1b[1] + 256;
    }

    // ---- problem constants ----
    const int Bn = 4, n1 = 2048, m1 = 1024, m2 = 512, KR = 64;
    const int R1 = Bn * m1 * KR;   // 262144 edge rows (SA1)
    const int R2 = Bn * m2 * KR;   // 131072 edge rows (SA2)
    const int NP = Bn * n1;        // 8192 points

    // ---- workspace layout (bump allocator) ----
    char*  base = (char*)d_ws;
    size_t off  = 0;
    auto alloc = [&](size_t bytes) -> void* {
        void* p = base + off;
        off = (off + bytes + 255) & ~(size_t)255;
        return p;
    };
    __bf16* bb0   = (__bf16*)alloc((size_t)R1 * 160 * 2);  // 84 MB ping
    __bf16* bb1   = (__bf16*)alloc((size_t)R1 * 256 * 2);  // 134 MB pong
    __bf16* wtbuf = (__bf16*)alloc((size_t)256 * 512 * 2); // transposed weights
    float*  x0    = (float*)alloc((size_t)NP * 128 * 4);
    float*  x1    = (float*)alloc((size_t)Bn * m1 * 256 * 4);
    float*  h1    = (float*)alloc((size_t)Bn * m1 * 256 * 4);
    float*  h2    = (float*)alloc((size_t)Bn * m2 * 256 * 4);
    float*  x2    = (float*)alloc((size_t)Bn * m2 * 256 * 4);
    float*  xi    = (float*)alloc((size_t)Bn * m1 * 256 * 4);
    float*  xf2m  = (float*)alloc((size_t)Bn * m1 * 256 * 4);
    float*  xf2   = (float*)alloc((size_t)Bn * m1 * 256 * 4);
    float*  xi1   = (float*)alloc((size_t)NP * 256 * 4);
    float*  posq1 = (float*)alloc((size_t)Bn * m1 * 3 * 4);
    float*  posq2 = (float*)alloc((size_t)Bn * m2 * 3 * 4);
    float*  stats = (float*)alloc(513 * 4);
    int*    idx1  = (int*)alloc((size_t)Bn * m1 * 4);
    int*    idx2  = (int*)alloc((size_t)Bn * m2 * 4);
    int*    nbr1  = (int*)alloc((size_t)R1 * 4);
    int*    mask1 = (int*)alloc((size_t)R1 * 4);
    int*    nbr2  = (int*)alloc((size_t)R2 * 4);
    int*    mask2 = (int*)alloc((size_t)R2 * 4);

    // ---- helpers ----
    auto gemm = [&](const __bf16* A, int Kpad, const float* W, int N, int Kdim,
                    const float* bias, float* Cf, __bf16* Cb, int ldc, int M) {
        const int wtot = N * Kpad;
        transpose_w_kernel<<<(wtot + 255) / 256, 256, 0, stream>>>(W, Kdim, N, Kpad, wtbuf);
        dim3 g(M / 64, N / 64);
        wmma_gemm_kernel<<<g, 256, 0, stream>>>(A, Kpad, wtbuf, bias, Cf, Cb, ldc, M);
    };
    auto bn = [&](__bf16* H, int N, int M, const int* mask) {
        zero_f32_kernel<<<(2 * N + 1 + 255) / 256, 256, 0, stream>>>(stats, 2 * N + 1);
        bn_stats_kernel<<<1024, 256, 0, stream>>>(H, N, mask, M, N, stats);
        bn_norm_relu_kernel<<<8192, 256, 0, stream>>>(H, N, M, N, stats);
    };

    // ==== 1. x0 = Linear([emb|prompt]) ====
    build_x0_in_kernel<<<2048, 256, 0, stream>>>(x_tok, index, emb, pemb, n1, 160, bb0, Bn);
    gemm(bb0, 160, W1, 128, 136, b1, x0, nullptr, 128, NP);

    // ==== 2. Set abstraction 1 (n=2048 -> m=1024, r=0.2) ====
    fps_kernel<<<Bn, 256, 0, stream>>>(pos, n1, m1, idx1);
    gather_posq_kernel<<<(Bn * m1 + 255) / 256, 256, 0, stream>>>(pos, idx1, n1, m1, Bn, posq1);
    knn_query_kernel<<<Bn * m1, 256, 0, stream>>>(pos, posq1, n1, m1, KR, 0.04f, nbr1, mask1);
    gather_msg_kernel<<<8192, 256, 0, stream>>>(x0, 128, pos, posq1, nbr1, n1, m1, KR, 160, bb0, Bn);
    gemm(bb0, 160, sa1W[0], 128, 131, sa1b[0], nullptr, bb1, 128, R1);
    bn(bb1, 128, R1, mask1);
    gemm(bb1, 128, sa1W[1], 128, 128, sa1b[1], nullptr, bb0, 128, R1);
    bn(bb0, 128, R1, mask1);
    gemm(bb0, 128, sa1W[2], 256, 128, sa1b[2], nullptr, bb1, 256, R1);
    masked_max_kernel<<<Bn * m1, 256, 0, stream>>>(bb1, 256, mask1, KR, 256, h1);

    // ==== 3. x1 = Linear([h1|prompt]) ====
    concat_prompt_kernel<<<2048, 256, 0, stream>>>(h1, 256, index, pemb, m1, 288, bb0, Bn);
    gemm(bb0, 288, W2, 256, 264, b2, x1, nullptr, 256, Bn * m1);

    // ==== 4. Set abstraction 2 (n=1024 -> m=512, r=0.4) ====
    fps_kernel<<<Bn, 256, 0, stream>>>(posq1, m1, m2, idx2);
    gather_posq_kernel<<<(Bn * m2 + 255) / 256, 256, 0, stream>>>(posq1, idx2, m1, m2, Bn, posq2);
    knn_query_kernel<<<Bn * m2, 256, 0, stream>>>(posq1, posq2, m1, m2, KR, 0.16f, nbr2, mask2);
    gather_msg_kernel<<<8192, 256, 0, stream>>>(x1, 256, posq1, posq2, nbr2, m1, m2, KR, 288, bb0, Bn);
    gemm(bb0, 288, sa2W[0], 256, 259, sa2b[0], nullptr, bb1, 256, R2);
    bn(bb1, 256, R2, mask2);
    gemm(bb1, 256, sa2W[1], 256, 256, sa2b[1], nullptr, bb0, 256, R2);
    bn(bb0, 256, R2, mask2);
    gemm(bb0, 256, sa2W[2], 256, 256, sa2b[2], nullptr, bb1, 256, R2);
    masked_max_kernel<<<Bn * m2, 256, 0, stream>>>(bb1, 256, mask2, KR, 256, h2);

    // ==== 5. x2 = Linear([h2|prompt]) ====
    concat_prompt_kernel<<<1024, 256, 0, stream>>>(h2, 256, index, pemb, m2, 288, bb0, Bn);
    gemm(bb0, 288, W3, 256, 264, b3, x2, nullptr, 256, Bn * m2);

    // ==== 6. xi = knn_interpolate(x2, pos2 -> pos1) ====
    knn_interp3_kernel<<<Bn * m1, 256, 0, stream>>>(x2, posq2, posq1, m2, m1, 256, xi);

    // ==== 7. fp2 MLP([xi|x1]) ====
    concat2_kernel<<<4096, 256, 0, stream>>>(xi, 256, x1, 256, Bn * m1, 512, bb0);
    gemm(bb0, 512, fp2W[0], 256, 512, fp2b[0], nullptr, bb1, 256, Bn * m1);
    bn(bb1, 256, Bn * m1, nullptr);
    gemm(bb1, 256, fp2W[1], 256, 256, fp2b[1], xf2m, nullptr, 256, Bn * m1);

    // ==== 8. xf2 = Linear([fp2_out|prompt]) ====
    concat_prompt_kernel<<<2048, 256, 0, stream>>>(xf2m, 256, index, pemb, m1, 288, bb0, Bn);
    gemm(bb0, 288, W4, 256, 264, b4, xf2, nullptr, 256, Bn * m1);

    // ==== 9. xi1 = knn_interpolate(xf2, pos1 -> pos) ====
    knn_interp3_kernel<<<NP, 256, 0, stream>>>(xf2, posq1, pos, m1, n1, 256, xi1);

    // ==== 10. fp1 MLP([xi1|x0]) -> output ====
    concat2_kernel<<<8192, 256, 0, stream>>>(xi1, 256, x0, 128, NP, 384, bb0);
    gemm(bb0, 384, fp1W[0], 256, 384, fp1b[0], nullptr, bb1, 256, NP);
    bn(bb1, 256, NP, nullptr);
    gemm(bb1, 256, fp1W[1], 256, 256, fp1b[1], nullptr, bb0, 256, NP);
    bn(bb0, 256, NP, nullptr);
    gemm(bb0, 256, fp1W[2], 128, 256, fp1b[2], (float*)d_out, nullptr, 128, NP);
}
